// EENHPool_31533649887814
// MI455X (gfx1250) — matrix operand order repaired and checked by reading, not compiled, not run on hardware
//
#include <hip/hip_runtime.h>
#include <hip/hip_fp16.h>
#include <math.h>

typedef __attribute__((ext_vector_type(16))) _Float16 v16h;
typedef __attribute__((ext_vector_type(8)))  float    v8f;

#define NA      50000
#define NB      50000
#define EE      320000
#define INDIM   128
#define EDGEDIM 64
#define HID     128

// ---- order-preserving float <-> uint mapping for atomicMax on scores ----
__device__ __forceinline__ unsigned fmap(float f) {
  unsigned u = __float_as_uint(f);
  return (u & 0x80000000u) ? ~u : (u | 0x80000000u);
}
__device__ __forceinline__ float funmap(unsigned u) {
  return (u & 0x80000000u) ? __uint_as_float(u & 0x7fffffffu)
                           : __uint_as_float(~u);
}

#define CVT4(dst, b, f4)                 \
  dst[(b) + 0] = (_Float16)(f4).x;       \
  dst[(b) + 1] = (_Float16)(f4).y;       \
  dst[(b) + 2] = (_Float16)(f4).z;       \
  dst[(b) + 3] = (_Float16)(f4).w;

// Load one A fragment (16x32 f16 tile) from an f32 row-major matrix.
// Lane l holds row (l&15); half i maps to k = ((i<8)?i:i+8) + 8*(l>>4) + kbase,
// i.e. two runs of 8 consecutive k -> four float4 loads.
__device__ __forceinline__ v16h load_a_frag(const float4* row4, int hi, int kbase) {
  const int k0 = kbase + hi * 8;        // multiple of 8
  float4 r0 = row4[(k0 >> 2) + 0];
  float4 r1 = row4[(k0 >> 2) + 1];
  float4 r2 = row4[((k0 + 16) >> 2) + 0];
  float4 r3 = row4[((k0 + 16) >> 2) + 1];
  v16h a;
  CVT4(a, 0, r0); CVT4(a, 4, r1); CVT4(a, 8, r2); CVT4(a, 12, r3);
  return a;
}

// =====================================================================
// Pack an f32 weight W[Ktot][128] into f16 B-fragments:
// frag[((q*8+nb)*32 + lane)*16 + i] = f16(W[k][n]),
//   k = ((i<8)?i:i+8) + 8*(lane>>4) + 32*q, n = nb*16 + (lane&15)
// so a wave reads its whole B fragment with 2x global_load_b128.
// =====================================================================
__global__ void k_pack_w(const float* __restrict__ W, _Float16* __restrict__ dst,
                         int ktot) {
  const int tid = blockIdx.x * blockDim.x + threadIdx.x;
  if (tid >= ktot * 128) return;
  const int i  = tid & 15;
  const int l  = (tid >> 4) & 31;
  const int nb = (tid >> 9) & 7;
  const int q  = tid >> 12;
  const int k  = ((i < 8) ? i : i + 8) + 8 * (l >> 4) + 32 * q;
  const int n  = nb * 16 + (l & 15);
  dst[tid] = (_Float16)W[k * 128 + n];
}

// =====================================================================
// Node projections with packed-f16 weights.
// grid.y==0: X=x_a -> P_a_src (Wn_src) AND P_a_dst (Wn_dst)  (A reused)
// grid.y==1: X=x_b -> P_b_dst (Wn_dst)
// =====================================================================
__device__ __forceinline__ void proj_tiles(const v16h* a, const v16h* __restrict__ Bf,
                                           float* __restrict__ P, int n0, int lane,
                                           int hi) {
#pragma unroll 1
  for (int nb = 0; nb < 8; ++nb) {
    const int n = nb * 16 + (lane & 15);
    v8f c = {};
#pragma unroll
    for (int q = 0; q < 4; ++q) {
      v16h b = Bf[(q * 8 + nb) * 32 + lane];
      c = __builtin_amdgcn_wmma_f32_16x16x32_f16(false, a[q], false, b,
                                                 (short)0, c, false, false);
    }
#pragma unroll
    for (int r = 0; r < 8; ++r)
      P[(size_t)(n0 + r + hi * 8) * HID + n] = c[r];
  }
}

__global__ void k_proj(const float* __restrict__ x_a, const float* __restrict__ x_b,
                       const _Float16* __restrict__ wn_src_h,
                       const _Float16* __restrict__ wn_dst_h,
                       float* __restrict__ P_a_src, float* __restrict__ P_a_dst,
                       float* __restrict__ P_b_dst) {
  const int wave = (blockIdx.x * blockDim.x + threadIdx.x) >> 5;
  const int lane = threadIdx.x & 31;
  const int NT = NA / 16; // 3125
  if (wave >= NT) return;
  const int n0 = wave * 16;
  const int m  = lane & 15;
  const int hi = lane >> 4;
  const float* X = (blockIdx.y == 0) ? x_a : x_b;
  const float4* row4 = (const float4*)(X + (size_t)(n0 + m) * INDIM);

  v16h a[4];
#pragma unroll
  for (int q = 0; q < 4; ++q) a[q] = load_a_frag(row4, hi, q * 32);

  if (blockIdx.y == 0) {
    proj_tiles(a, (const v16h*)wn_src_h, P_a_src, n0, lane, hi);
    proj_tiles(a, (const v16h*)wn_dst_h, P_a_dst, n0, lane, hi);
  } else {
    proj_tiles(a, (const v16h*)wn_dst_h, P_b_dst, n0, lane, hi);
  }
}

// =====================================================================
// Per-edge: h = relu(ea@Wh + Psrc[s] + Pdst[d] + (Wh_b+Wn_b)); score = h.w_e
// one wave per 16 edges; K=64 => 2 WMMAs per 16x16 tile, 8 tiles across HID
// =====================================================================
__global__ void k_edge_scores(const float* __restrict__ ea, const int* __restrict__ ei,
                              const _Float16* __restrict__ wh_h,
                              const float* __restrict__ Wh_b,
                              const float* __restrict__ Wn_b, const float* __restrict__ w_e,
                              const float* __restrict__ Psrc, const float* __restrict__ Pdst,
                              float* __restrict__ scores) {
  const int wave = (blockIdx.x * blockDim.x + threadIdx.x) >> 5;
  const int lane = threadIdx.x & 31;
  const int NT = EE / 16; // 20000
  if (wave >= NT) return;
  const int e0 = wave * 16;
  const int m  = lane & 15;
  const int hi = lane >> 4;
  const v16h* Bf = (const v16h*)wh_h;

  const float4* row4 = (const float4*)(ea + (size_t)(e0 + m) * EDGEDIM);
  v16h a0 = load_a_frag(row4, hi, 0);
  v16h a1 = load_a_frag(row4, hi, 32);

  int sidx[8], didx[8];
#pragma unroll
  for (int r = 0; r < 8; ++r) {
    int mm = e0 + r + hi * 8;
    sidx[r] = ei[mm];
    didx[r] = ei[EE + mm];
  }
  float acc[8] = {0.f, 0.f, 0.f, 0.f, 0.f, 0.f, 0.f, 0.f};
#pragma unroll 1
  for (int nb = 0; nb < 8; ++nb) {
    const int n = nb * 16 + (lane & 15);
    v16h b0 = Bf[(0 * 8 + nb) * 32 + lane];
    v16h b1 = Bf[(1 * 8 + nb) * 32 + lane];
    v8f c = {};
    c = __builtin_amdgcn_wmma_f32_16x16x32_f16(false, a0, false, b0, (short)0, c, false, false);
    c = __builtin_amdgcn_wmma_f32_16x16x32_f16(false, a1, false, b1, (short)0, c, false, false);
    const float bias = Wh_b[n] + Wn_b[n];
    const float we = w_e[n];
#pragma unroll
    for (int r = 0; r < 8; ++r) {
      float h = c[r] + Psrc[(size_t)sidx[r] * HID + n]
                     + Pdst[(size_t)didx[r] * HID + n] + bias;
      acc[r] += we * fmaxf(h, 0.0f);
    }
  }
  // reduce across the 16 lanes holding the same row (xor masks stay in-half)
#pragma unroll
  for (int r = 0; r < 8; ++r) {
    float v = acc[r];
    v += __shfl_xor(v, 1, 32);
    v += __shfl_xor(v, 2, 32);
    v += __shfl_xor(v, 4, 32);
    v += __shfl_xor(v, 8, 32);
    acc[r] = v;
  }
  if ((lane & 15) == 0) {
#pragma unroll
    for (int r = 0; r < 8; ++r)
      scores[e0 + r + hi * 8] = acc[r];
  }
}

// =====================================================================
// softmax reductions
// =====================================================================
__global__ void k_init_slots(unsigned* slots) {
  if (blockIdx.x == 0 && threadIdx.x < 8) slots[threadIdx.x] = 0u;
}

__global__ void k_reduce_max(const float* __restrict__ s, int n, unsigned* slot) {
  float m = -3.4e38f;
  for (int i = blockIdx.x * blockDim.x + threadIdx.x; i < n; i += gridDim.x * blockDim.x)
    m = fmaxf(m, s[i]);
#pragma unroll
  for (int off = 16; off >= 1; off >>= 1) m = fmaxf(m, __shfl_xor(m, off, 32));
  __shared__ float sm[8];
  if ((threadIdx.x & 31) == 0) sm[threadIdx.x >> 5] = m;
  __syncthreads();
  if (threadIdx.x == 0) {
    for (int i = 1; i < (int)(blockDim.x >> 5); ++i) m = fmaxf(m, sm[i]);
    atomicMax(slot, fmap(m));
  }
}

__global__ void k_reduce_sumexp(const float* __restrict__ s, int n,
                                const unsigned* __restrict__ maxslot,
                                float* sumslot) {
  const float smax = funmap(*maxslot);
  float acc = 0.f;
  for (int i = blockIdx.x * blockDim.x + threadIdx.x; i < n; i += gridDim.x * blockDim.x)
    acc += expf(s[i] - smax);
#pragma unroll
  for (int off = 16; off >= 1; off >>= 1) acc += __shfl_xor(acc, off, 32);
  __shared__ float sm[8];
  if ((threadIdx.x & 31) == 0) sm[threadIdx.x >> 5] = acc;
  __syncthreads();
  if (threadIdx.x == 0) {
    for (int i = 1; i < (int)(blockDim.x >> 5); ++i) acc += sm[i];
    atomicAdd(sumslot, acc);
  }
}

// =====================================================================
// acc_a = x_a, acc_b = x_b  (so acc ends as x + lf after scatter)
// =====================================================================
__global__ void k_init_acc(const float4* __restrict__ x_a, const float4* __restrict__ x_b,
                           float4* __restrict__ acc_a, float4* __restrict__ acc_b) {
  const size_t n = (size_t)NA * INDIM / 4;
  for (size_t i = blockIdx.x * (size_t)blockDim.x + threadIdx.x; i < n;
       i += (size_t)gridDim.x * blockDim.x) {
    acc_a[i] = x_a[i];
    acc_b[i] = x_b[i];
  }
}

// =====================================================================
// scatter: one wave per edge; 4 cols per lane; fp32 global atomics (L2-served)
// =====================================================================
__global__ void k_scatter_ab(const float* __restrict__ x_a, const float* __restrict__ x_b,
                             const int* __restrict__ ei, const float* __restrict__ scores,
                             const unsigned* __restrict__ maxslot,
                             const float* __restrict__ sumslot,
                             float* __restrict__ acc_a, float* __restrict__ acc_b,
                             float* __restrict__ alpha_out) {
  const int e = (blockIdx.x * blockDim.x + threadIdx.x) >> 5;
  const int lane = threadIdx.x & 31;
  if (e >= EE) return;
  const float alpha = expf(scores[e] - funmap(*maxslot)) / (*sumslot);
  if (lane == 0) alpha_out[e] = alpha;
  const size_t s = (size_t)ei[e] * INDIM;
  const size_t d = (size_t)ei[EE + e] * INDIM;
#pragma unroll
  for (int j = 0; j < 4; ++j) {
    const int c = lane + 32 * j;
    const float xav = x_a[s + c];
    const float xbv = x_b[d + c];
    atomicAdd(&acc_a[s + c], -alpha * xbv);
    atomicAdd(&acc_b[d + c], -alpha * xav);
  }
}

__global__ void k_scatter_aa(const float* __restrict__ x_a,
                             const int* __restrict__ ei, const float* __restrict__ scores,
                             const unsigned* __restrict__ maxslot,
                             const float* __restrict__ sumslot,
                             float* __restrict__ acc_a,
                             float* __restrict__ alpha_out) {
  const int e = (blockIdx.x * blockDim.x + threadIdx.x) >> 5;
  const int lane = threadIdx.x & 31;
  if (e >= EE) return;
  const float alpha = expf(scores[e] - funmap(*maxslot)) / (*sumslot);
  if (lane == 0) alpha_out[e] = alpha;
  const size_t s = (size_t)ei[e] * INDIM;
  const size_t d = (size_t)ei[EE + e] * INDIM;
#pragma unroll
  for (int j = 0; j < 4; ++j) {
    const int c = lane + 32 * j;
    atomicAdd(&acc_a[s + c], -alpha * x_a[d + c]);
  }
}

// =====================================================================
// out = t + t@ft_w + ft_b  (t = acc), packed-f16 weights, fp32 residual
// =====================================================================
__global__ void k_final(const float* __restrict__ T, const _Float16* __restrict__ ft_h,
                        const float* __restrict__ ft_b, float* __restrict__ out) {
  const int wave = (blockIdx.x * blockDim.x + threadIdx.x) >> 5;
  const int lane = threadIdx.x & 31;
  const int NT = NA / 16;
  if (wave >= NT) return;
  const int n0 = wave * 16;
  const int m  = lane & 15;
  const int hi = lane >> 4;
  const v16h* Bf = (const v16h*)ft_h;
  const float4* row4 = (const float4*)(T + (size_t)(n0 + m) * INDIM);

  v16h a[4];
#pragma unroll
  for (int q = 0; q < 4; ++q) a[q] = load_a_frag(row4, hi, q * 32);

#pragma unroll 1
  for (int nb = 0; nb < 8; ++nb) {
    const int n = nb * 16 + (lane & 15);
    v8f c = {};
#pragma unroll
    for (int q = 0; q < 4; ++q) {
      v16h b = Bf[(q * 8 + nb) * 32 + lane];
      c = __builtin_amdgcn_wmma_f32_16x16x32_f16(false, a[q], false, b,
                                                 (short)0, c, false, false);
    }
    const float bb = ft_b[n];
#pragma unroll
    for (int r = 0; r < 8; ++r) {
      const size_t idx = (size_t)(n0 + r + hi * 8) * INDIM + n;
      out[idx] = T[idx] + c[r] + bb;
    }
  }
}

// =====================================================================
extern "C" void kernel_launch(void* const* d_in, const int* in_sizes, int n_in,
                              void* d_out, int out_size, void* d_ws, size_t ws_size,
                              hipStream_t stream) {
  (void)in_sizes; (void)n_in; (void)out_size; (void)ws_size;
  const float* x_a   = (const float*)d_in[0];
  const float* x_b   = (const float*)d_in[1];
  const int*   ei_ab = (const int*)  d_in[2];
  const float* ea_ab = (const float*)d_in[3];
  const int*   ei_aa = (const int*)  d_in[4];
  const float* ea_aa = (const float*)d_in[5];
  const float* Wh_w  = (const float*)d_in[6];
  const float* Wh_b  = (const float*)d_in[7];
  const float* Wn_w  = (const float*)d_in[8];
  const float* Wn_b  = (const float*)d_in[9];
  const float* w_e   = (const float*)d_in[10];
  const float* ft_w  = (const float*)d_in[11];
  const float* ft_b  = (const float*)d_in[12];

  const size_t NODE = (size_t)NA * INDIM; // 6.4M floats
  float* base    = (float*)d_ws;
  float* P_a_src = base;            // later aliased as acc_a
  float* P_a_dst = base + NODE;     // later aliased as acc_b
  float* P_b_dst = base + 2 * NODE;
  float* sc_ab   = base + 3 * NODE;
  float* sc_aa   = sc_ab + EE;
  unsigned* slots = (unsigned*)(sc_aa + EE); // [0]maxAB [1]maxAA [2]sumAB [3]sumAA
  // packed f16 weights (32B-aligned: all offsets are multiples of 8 floats)
  _Float16* wh_h     = (_Float16*)(sc_aa + EE + 8);      //  64*128 halfs
  _Float16* wn_src_h = wh_h + 64 * 128;                  // 128*128 halfs
  _Float16* wn_dst_h = wn_src_h + 128 * 128;             // 128*128 halfs
  _Float16* ft_h     = wn_dst_h + 128 * 128;             // 128*128 halfs

  float* out_a = (float*)d_out;
  float* out_b = out_a + NODE;
  float* al_ab = out_b + NODE;
  float* al_aa = al_ab + EE;

  dim3 blk(256);
  k_init_slots<<<1, 32, 0, stream>>>(slots);
  k_pack_w<<<32, blk, 0, stream>>>(Wh_w, wh_h, 64);
  k_pack_w<<<64, blk, 0, stream>>>(Wn_w, wn_src_h, 128);
  k_pack_w<<<64, blk, 0, stream>>>(Wn_w + INDIM * HID, wn_dst_h, 128);
  k_pack_w<<<64, blk, 0, stream>>>(ft_w, ft_h, 128);

  k_proj<<<dim3(391, 2), blk, 0, stream>>>(x_a, x_b, wn_src_h, wn_dst_h,
                                           P_a_src, P_a_dst, P_b_dst);
  k_edge_scores<<<2500, blk, 0, stream>>>(ea_ab, ei_ab, wh_h, Wh_b, Wn_b, w_e,
                                          P_a_src, P_b_dst, sc_ab);
  k_edge_scores<<<2500, blk, 0, stream>>>(ea_aa, ei_aa, wh_h, Wh_b, Wn_b, w_e,
                                          P_a_src, P_a_dst, sc_aa);
  k_reduce_max<<<512, blk, 0, stream>>>(sc_ab, EE, slots + 0);
  k_reduce_max<<<512, blk, 0, stream>>>(sc_aa, EE, slots + 1);
  k_reduce_sumexp<<<512, blk, 0, stream>>>(sc_ab, EE, slots + 0, (float*)(slots + 2));
  k_reduce_sumexp<<<512, blk, 0, stream>>>(sc_aa, EE, slots + 1, (float*)(slots + 3));
  // projections no longer needed -> reuse as accumulators (= x + lf)
  k_init_acc<<<1024, blk, 0, stream>>>((const float4*)x_a, (const float4*)x_b,
                                       (float4*)P_a_src, (float4*)P_a_dst);
  k_scatter_ab<<<40000, blk, 0, stream>>>(x_a, x_b, ei_ab, sc_ab, slots + 0,
                                          (const float*)(slots + 2),
                                          P_a_src, P_a_dst, al_ab);
  k_scatter_aa<<<40000, blk, 0, stream>>>(x_a, ei_aa, sc_aa, slots + 1,
                                          (const float*)(slots + 3),
                                          P_a_src, al_aa);
  k_final<<<391, blk, 0, stream>>>(P_a_src, ft_h, ft_b, out_a);
  k_final<<<391, blk, 0, stream>>>(P_a_dst, ft_h, ft_b, out_b);
}
